// Decoder_23055384445021
// MI455X (gfx1250) — compile-verified, matrix operand order checked by reading, not run on hardware
//
#include <hip/hip_runtime.h>

typedef __attribute__((ext_vector_type(16))) __bf16   v16bf;
typedef __attribute__((ext_vector_type(8)))  __bf16   bf16x8;
typedef __attribute__((ext_vector_type(8)))  float    v8f;
typedef __attribute__((ext_vector_type(4)))  unsigned int u32x4;

union ABu { v16bf v; u32x4 u[2]; unsigned short s[16]; };

__device__ inline unsigned short f2bf(float f) {
  union { float f; unsigned u; } c; c.f = f;
  unsigned u = c.u;
  return (unsigned short)((u + 0x7fffu + ((u >> 16) & 1u)) >> 16);
}

// branch-free transcendental helpers (v_exp_f32 / v_rcp_f32, no EXEC divergence)
__device__ inline float fexp(float x) {
  return __builtin_amdgcn_exp2f(x * 1.4426950408889634f);
}
__device__ inline float frcp(float x) { return __builtin_amdgcn_rcpf(x); }
__device__ inline float fast_sigmoid(float x) {
  x = fminf(fmaxf(x, -30.f), 30.f);
  return frcp(1.f + fexp(-x));
}
__device__ inline float fast_tanh(float x) {
  x = fminf(fmaxf(x, -15.f), 15.f);
  float t = fexp(2.f * x);
  return (t - 1.f) * frcp(t + 1.f);
}

// dot( bf16 weight row (K elems, 16B aligned), f32 vector in LDS )
// (float)__bf16 * float + acc -> v_fma_mix_f32_bf16 / v_cvt+fma
__device__ inline float dotw(const unsigned short* __restrict__ w,
                             const float* __restrict__ x, int K) {
  float acc = 0.f;
  const bf16x8* wp = (const bf16x8*)w;
  for (int k = 0; k < K; k += 8) {
    bf16x8 p = *wp++;
    const float* xx = x + k;
#pragma unroll
    for (int j = 0; j < 8; ++j) acc = fmaf((float)p[j], xx[j], acc);
  }
  return acc;
}

// ---------------- weight transpose f32 (K,N) -> bf16 (N,Kpad) ----------------
__global__ __launch_bounds__(256) void transpose_bf16_kernel(
    const float* __restrict__ W, unsigned short* __restrict__ Wt,
    int K, int N, int Kpad) {
  int i = blockIdx.x * 256 + threadIdx.x;
  if (i >= N * Kpad) return;
  int n = i / Kpad, k = i % Kpad;
  float v = (k < K) ? W[(size_t)k * N + n] : 0.f;
  Wt[i] = f2bf(v);
}

// ---------------- prenet input: (B*400, 96) bf16, K padded 80->96 ----------------
__global__ __launch_bounds__(256) void build_prenet_kernel(
    const float* __restrict__ memory, unsigned short* __restrict__ X) {
  int i = blockIdx.x * 256 + threadIdx.x;
  const int TOT = 64 * 400 * 96;
  if (i >= TOT) return;
  int r = i / 96, k = i % 96;
  int b = r / 400, s = r % 400;
  float v = 0.f;
  if (s > 0 && k < 80) v = memory[((size_t)b * 800 + (2 * s - 1)) * 80 + k];
  X[i] = f2bf(v);
}

// ---------------- generic WMMA GEMM: C(M,N) = act(A(M,K) @ Bt(N,K)^T + bias) ----------------
template <int AF32, int RELU, int OBF16>
__global__ __launch_bounds__(256) void wmma_gemm_kernel(
    const void* __restrict__ A, const unsigned short* __restrict__ Bt,
    const float* __restrict__ bias, void* __restrict__ C,
    int M, int N, int K, int lda, int ldb, int ldc) {
  const int wv  = threadIdx.x >> 5;
  const int lane = threadIdx.x & 31;
  const int tilesN = N >> 4;
  const int tile = blockIdx.x * 8 + wv;
  if (tile >= (M >> 4) * tilesN) return;
  const int tm = tile / tilesN, tn = tile % tilesN;
  const int hlf = lane >> 4, l15 = lane & 15;
  const int kb = hlf * 8;
  v8f c = {0.f, 0.f, 0.f, 0.f, 0.f, 0.f, 0.f, 0.f};
  for (int k0 = 0; k0 < K; k0 += 32) {
    ABu a, bm;
    if (AF32) {
      const float* ap = (const float*)A + (size_t)(tm * 16 + l15) * lda + k0 + kb;
#pragma unroll
      for (int j = 0; j < 8; ++j) a.s[j] = f2bf(ap[j]);
#pragma unroll
      for (int j = 0; j < 8; ++j) a.s[8 + j] = f2bf(ap[16 + j]);
    } else {
      const unsigned short* ap =
          (const unsigned short*)A + (size_t)(tm * 16 + l15) * lda + k0 + kb;
      a.u[0] = *(const u32x4*)(ap);
      a.u[1] = *(const u32x4*)(ap + 16);
    }
    const unsigned short* bp = Bt + (size_t)(tn * 16 + l15) * ldb + k0 + kb;
    bm.u[0] = *(const u32x4*)(bp);
    bm.u[1] = *(const u32x4*)(bp + 16);
    c = __builtin_amdgcn_wmma_f32_16x16x32_bf16(false, a.v, false, bm.v,
                                                (short)0, c, false, false);
  }
  const int n = tn * 16 + l15;
  const float bv = bias ? bias[n] : 0.f;
#pragma unroll
  for (int v = 0; v < 8; ++v) {
    int m = tm * 16 + v + hlf * 8;
    float o = c[v] + bv;
    if (RELU) o = fmaxf(o, 0.f);
    if (OBF16) ((unsigned short*)C)[(size_t)m * ldc + n] = f2bf(o);
    else       ((float*)C)[(size_t)m * ldc + n] = o;
  }
}

// ---------------- persistent per-batch decoder (64 blocks x 256 threads) ----------------
__global__ __launch_bounds__(256) void decoder_kernel(
    const float* __restrict__ inputs, const unsigned char* __restrict__ mask,
    const float* __restrict__ pv, const float* __restrict__ x_all,
    const unsigned short* __restrict__ ak_t, const unsigned short* __restrict__ ark_t,
    const float* __restrict__ attn_b,
    const unsigned short* __restrict__ Wq_t, const unsigned short* __restrict__ Wloc_t,
    const float* __restrict__ v_w, const float* __restrict__ v_b,
    const float* __restrict__ conv_w,
    const unsigned short* __restrict__ proj_t, const float* __restrict__ proj_b,
    const unsigned short* __restrict__ d1k_t, const unsigned short* __restrict__ d1rk_t,
    const float* __restrict__ d1b,
    const unsigned short* __restrict__ d2k_t, const unsigned short* __restrict__ d2rk_t,
    const float* __restrict__ d2b,
    const unsigned short* __restrict__ mel_t, const float* __restrict__ mel_b,
    const float* __restrict__ stop_w, const float* __restrict__ stop_b,
    float* __restrict__ outs_out, float* __restrict__ attns_out,
    float* __restrict__ stops_out) {
  __shared__ float s_cat[768];
  __shared__ float s_gx[768], s_gh[768];
  __shared__ float s_h[256], s_d1[256], s_d2[256];
  __shared__ float s_ctx[512], s_q[128];
  __shared__ float s_attn[512], s_cum[512], s_score[512];
  __shared__ __align__(16) unsigned short s_wloc[128 * 32];
  __shared__ float s_vw[128];
  __shared__ float s_cw[31 * 64];
  __shared__ float s_din[256], s_din2[256], s_din3[256];
  __shared__ float s_ob[160];
  __shared__ float s_red[256];

  const int b = blockIdx.x;
  const int tid = threadIdx.x;
  const int wv = tid >> 5, lane = tid & 31;
  const int hlf = lane >> 4, l15 = lane & 15;
  const int kb = hlf * 8;
  const float* pvb = pv + (size_t)b * 512 * 128;
  const float* inb = inputs + (size_t)b * 512 * 512;
  const unsigned char* mb = mask + (size_t)b * 512;
  const float vb0 = v_b[0];
  const float sb0 = stop_b[0];

  for (int i = tid; i < 512; i += 256) { s_attn[i] = 0.f; s_cum[i] = 0.f; s_ctx[i] = 0.f; }
  s_h[tid] = 0.f; s_d1[tid] = 0.f; s_d2[tid] = 0.f;
  for (int i = tid; i < 128 * 32; i += 256) s_wloc[i] = Wloc_t[i];
  if (tid < 128) s_vw[tid] = v_w[tid];
  for (int i = tid; i < 31 * 64; i += 256) s_cw[i] = conv_w[i];
  __syncthreads();

  for (int s = 0; s < 400; ++s) {
    // 1) cat = [x_t(128), ctx(512)]
    if (tid < 128) s_cat[tid] = x_all[((size_t)b * 400 + s) * 128 + tid];
    for (int d = tid; d < 512; d += 256) s_cat[128 + d] = s_ctx[d];
    __syncthreads();
    // 2) attention GRU gates
    for (int j = tid; j < 768; j += 256) {
      s_gx[j] = attn_b[j]       + dotw(ak_t  + (size_t)j * 640, s_cat, 640);
      s_gh[j] = attn_b[768 + j] + dotw(ark_t + (size_t)j * 256, s_h, 256);
    }
    __syncthreads();
    // 3) GRU combine (z,r,h order, reset_after)
    {
      float z  = fast_sigmoid(s_gx[tid] + s_gh[tid]);
      float r  = fast_sigmoid(s_gx[256 + tid] + s_gh[256 + tid]);
      float hc = fast_tanh(s_gx[512 + tid] + r * s_gh[512 + tid]);
      s_h[tid] = z * s_h[tid] + (1.f - z) * hc;
    }
    __syncthreads();
    // 4) q = attn_h @ Wq
    if (tid < 128) s_q[tid] = dotw(Wq_t + (size_t)tid * 256, s_h, 256);
    __syncthreads();
    // 5) fused location-conv -> WMMA (loc_f @ Wloc) -> tanh -> score
    for (int rt = wv; rt < 32; rt += 8) {
      const int t0 = rt * 16;
      const int m = t0 + l15;  // this lane's row
      float af[16];
#pragma unroll
      for (int j = 0; j < 16; ++j) af[j] = 0.f;
      for (int k = 0; k < 31; ++k) {
        int ti = m + k - 15;
        if (ti >= 0 && ti < 512) {
          float a0 = s_attn[ti], a1 = s_cum[ti];
          const float* wk = s_cw + k * 64;
#pragma unroll
          for (int j = 0; j < 16; ++j) {
            int f = kb + (j < 8 ? j : j + 8);
            af[j] += a0 * wk[f] + a1 * wk[32 + f];
          }
        }
      }
      ABu a;
#pragma unroll
      for (int j = 0; j < 16; ++j) a.s[j] = f2bf(af[j]);
      float acc[8];
#pragma unroll
      for (int v = 0; v < 8; ++v) acc[v] = 0.f;
      for (int ct = 0; ct < 8; ++ct) {
        ABu bm;
        const unsigned short* bp = s_wloc + (ct * 16 + l15) * 32 + kb;
        bm.u[0] = *(const u32x4*)(bp);
        bm.u[1] = *(const u32x4*)(bp + 16);
        v8f c = {0.f, 0.f, 0.f, 0.f, 0.f, 0.f, 0.f, 0.f};
        c = __builtin_amdgcn_wmma_f32_16x16x32_bf16(false, a.v, false, bm.v,
                                                    (short)0, c, false, false);
        const int n = ct * 16 + l15;
        const float qn = s_q[n], vn = s_vw[n];
#pragma unroll
        for (int v = 0; v < 8; ++v) {
          int t = t0 + v + hlf * 8;
          float e = fast_tanh(c[v] + pvb[(size_t)t * 128 + n] + qn);
          acc[v] += vn * e;
        }
      }
#pragma unroll
      for (int v = 0; v < 8; ++v) {
        acc[v] += __shfl_xor(acc[v], 1, 32);
        acc[v] += __shfl_xor(acc[v], 2, 32);
        acc[v] += __shfl_xor(acc[v], 4, 32);
        acc[v] += __shfl_xor(acc[v], 8, 32);
      }
      if (l15 == 0) {
#pragma unroll
        for (int v = 0; v < 8; ++v) s_score[t0 + v + hlf * 8] = acc[v];
      }
    }
    __syncthreads();
    // 6) bias + mask + softmax
    for (int t = tid; t < 512; t += 256) {
      float sc = s_score[t] + vb0;
      if (!mb[t]) sc = -1.0e9f;
      s_score[t] = sc;
    }
    __syncthreads();
    s_red[tid] = fmaxf(s_score[tid], s_score[tid + 256]);
    __syncthreads();
    for (int off = 128; off > 0; off >>= 1) {
      if (tid < off) s_red[tid] = fmaxf(s_red[tid], s_red[tid + off]);
      __syncthreads();
    }
    const float mx = s_red[0];
    __syncthreads();
    {
      float e0 = fexp(s_score[tid] - mx);
      float e1 = fexp(s_score[tid + 256] - mx);
      s_attn[tid] = e0; s_attn[tid + 256] = e1;
      s_red[tid] = e0 + e1;
    }
    __syncthreads();
    for (int off = 128; off > 0; off >>= 1) {
      if (tid < off) s_red[tid] += s_red[tid + off];
      __syncthreads();
    }
    const float inv = 1.f / s_red[0];
    __syncthreads();
    for (int t = tid; t < 512; t += 256) {
      float aa = s_attn[t] * inv;
      s_attn[t] = aa;
      s_cum[t] += aa;
      attns_out[((size_t)b * 400 + s) * 512 + t] = aa;
    }
    __syncthreads();
    // 7) ctx = attn @ inputs[b]
    for (int d = tid; d < 512; d += 256) {
      float acc = 0.f;
      for (int t = 0; t < 512; ++t) acc += s_attn[t] * inb[(size_t)t * 512 + d];
      s_ctx[d] = acc;
    }
    __syncthreads();
    // 8) dec_in = [h, ctx] @ proj + b
    s_cat[tid] = s_h[tid];
    for (int d = tid; d < 512; d += 256) s_cat[256 + d] = s_ctx[d];
    __syncthreads();
    s_din[tid] = proj_b[tid] + dotw(proj_t + (size_t)tid * 768, s_cat, 768);
    __syncthreads();
    // 9) decoder GRU 1 + residual
    for (int j = tid; j < 768; j += 256) {
      s_gx[j] = d1b[j]       + dotw(d1k_t  + (size_t)j * 256, s_din, 256);
      s_gh[j] = d1b[768 + j] + dotw(d1rk_t + (size_t)j * 256, s_d1, 256);
    }
    __syncthreads();
    {
      float z  = fast_sigmoid(s_gx[tid] + s_gh[tid]);
      float r  = fast_sigmoid(s_gx[256 + tid] + s_gh[256 + tid]);
      float hc = fast_tanh(s_gx[512 + tid] + r * s_gh[512 + tid]);
      float nh = z * s_d1[tid] + (1.f - z) * hc;
      s_d1[tid] = nh;
      s_din2[tid] = nh + s_din[tid];
    }
    __syncthreads();
    // 10) decoder GRU 2 + residual
    for (int j = tid; j < 768; j += 256) {
      s_gx[j] = d2b[j]       + dotw(d2k_t  + (size_t)j * 256, s_din2, 256);
      s_gh[j] = d2b[768 + j] + dotw(d2rk_t + (size_t)j * 256, s_d2, 256);
    }
    __syncthreads();
    {
      float z  = fast_sigmoid(s_gx[tid] + s_gh[tid]);
      float r  = fast_sigmoid(s_gx[256 + tid] + s_gh[256 + tid]);
      float hc = fast_tanh(s_gx[512 + tid] + r * s_gh[512 + tid]);
      float nh = z * s_d2[tid] + (1.f - z) * hc;
      s_d2[tid] = nh;
      s_din3[tid] = nh + s_din2[tid];
    }
    __syncthreads();
    // 11) mel projection + stop token
    if (tid < 160) {
      float o = mel_b[tid] + dotw(mel_t + (size_t)tid * 256, s_din3, 256);
      s_ob[tid] = o;
      outs_out[((size_t)b * 400 + s) * 160 + tid] = o;
    }
    __syncthreads();
    {
      float p = s_din3[tid] * stop_w[tid];
      if (tid < 160) p += s_ob[tid] * stop_w[256 + tid];
      s_red[tid] = p;
    }
    __syncthreads();
    for (int off = 128; off > 0; off >>= 1) {
      if (tid < off) s_red[tid] += s_red[tid + off];
      __syncthreads();
    }
    if (tid == 0) stops_out[(size_t)b * 400 + s] = s_red[0] + sb0;
    __syncthreads();
  }
}

extern "C" void kernel_launch(void* const* d_in, const int* in_sizes, int n_in,
                              void* d_out, int out_size, void* d_ws, size_t ws_size,
                              hipStream_t stream) {
  const float* inputs    = (const float*)d_in[0];
  const float* memory    = (const float*)d_in[1];
  const unsigned char* mask = (const unsigned char*)d_in[2];
  const float* prenet_w1 = (const float*)d_in[3];
  const float* prenet_b1 = (const float*)d_in[4];
  const float* prenet_w2 = (const float*)d_in[5];
  const float* prenet_b2 = (const float*)d_in[6];
  const float* attn_k    = (const float*)d_in[7];
  const float* attn_rk   = (const float*)d_in[8];
  const float* attn_b    = (const float*)d_in[9];
  const float* Wq        = (const float*)d_in[10];
  const float* Wv        = (const float*)d_in[11];
  const float* v_w       = (const float*)d_in[12];
  const float* v_b       = (const float*)d_in[13];
  const float* conv_w    = (const float*)d_in[14];
  const float* Wloc      = (const float*)d_in[15];
  const float* proj_w    = (const float*)d_in[16];
  const float* proj_b    = (const float*)d_in[17];
  const float* d1k       = (const float*)d_in[18];
  const float* d1rk      = (const float*)d_in[19];
  const float* d1b       = (const float*)d_in[20];
  const float* d2k       = (const float*)d_in[21];
  const float* d2rk      = (const float*)d_in[22];
  const float* d2b       = (const float*)d_in[23];
  const float* mel_w     = (const float*)d_in[24];
  const float* mel_b     = (const float*)d_in[25];
  const float* stop_w    = (const float*)d_in[26];
  const float* stop_b    = (const float*)d_in[27];

  char* ws = (char*)d_ws;
  size_t off = 0;
  auto alloc = [&](size_t bytes) -> void* {
    void* p = ws + off;
    off += (bytes + 255) & ~(size_t)255;
    return p;
  };
  float* pvw   = (float*)alloc((size_t)64 * 512 * 128 * 4);
  float* x_all = (float*)alloc((size_t)64 * 400 * 128 * 4);
  unsigned short* pre_in = (unsigned short*)alloc((size_t)64 * 400 * 96 * 2);
  unsigned short* h1     = (unsigned short*)alloc((size_t)64 * 400 * 256 * 2);
  unsigned short* w1_t   = (unsigned short*)alloc((size_t)256 * 96 * 2);
  unsigned short* w2_t   = (unsigned short*)alloc((size_t)128 * 256 * 2);
  unsigned short* Wv_t   = (unsigned short*)alloc((size_t)128 * 512 * 2);
  unsigned short* ak_t   = (unsigned short*)alloc((size_t)768 * 640 * 2);
  unsigned short* ark_t  = (unsigned short*)alloc((size_t)768 * 256 * 2);
  unsigned short* Wq_t   = (unsigned short*)alloc((size_t)128 * 256 * 2);
  unsigned short* Wloc_t = (unsigned short*)alloc((size_t)128 * 32 * 2);
  unsigned short* proj_t = (unsigned short*)alloc((size_t)256 * 768 * 2);
  unsigned short* d1k_t  = (unsigned short*)alloc((size_t)768 * 256 * 2);
  unsigned short* d1rk_t = (unsigned short*)alloc((size_t)768 * 256 * 2);
  unsigned short* d2k_t  = (unsigned short*)alloc((size_t)768 * 256 * 2);
  unsigned short* d2rk_t = (unsigned short*)alloc((size_t)768 * 256 * 2);
  unsigned short* mel_t  = (unsigned short*)alloc((size_t)160 * 256 * 2);

  auto tr = [&](const float* W, unsigned short* Wt, int K, int N, int Kpad) {
    int tot = N * Kpad;
    transpose_bf16_kernel<<<(tot + 255) / 256, 256, 0, stream>>>(W, Wt, K, N, Kpad);
  };
  tr(prenet_w1, w1_t, 80, 256, 96);
  tr(prenet_w2, w2_t, 256, 128, 256);
  tr(Wv, Wv_t, 512, 128, 512);
  tr(attn_k, ak_t, 640, 768, 640);
  tr(attn_rk, ark_t, 256, 768, 256);
  tr(Wq, Wq_t, 256, 128, 256);
  tr(Wloc, Wloc_t, 32, 128, 32);
  tr(proj_w, proj_t, 768, 256, 768);
  tr(d1k, d1k_t, 256, 768, 256);
  tr(d1rk, d1rk_t, 256, 768, 256);
  tr(d2k, d2k_t, 256, 768, 256);
  tr(d2rk, d2rk_t, 256, 768, 256);
  tr(mel_w, mel_t, 256, 160, 256);

  {
    int tot = 64 * 400 * 96;
    build_prenet_kernel<<<(tot + 255) / 256, 256, 0, stream>>>(memory, pre_in);
  }
  {  // prenet layer 1: (25600,96) @ (96,256), relu, bf16 out
    int tiles = (25600 / 16) * (256 / 16);
    wmma_gemm_kernel<0, 1, 1><<<(tiles + 7) / 8, 256, 0, stream>>>(
        pre_in, w1_t, prenet_b1, h1, 25600, 256, 96, 96, 96, 256);
  }
  {  // prenet layer 2: (25600,256) @ (256,128), relu, f32 out
    int tiles = (25600 / 16) * (128 / 16);
    wmma_gemm_kernel<0, 1, 0><<<(tiles + 7) / 8, 256, 0, stream>>>(
        h1, w2_t, prenet_b2, x_all, 25600, 128, 256, 256, 256, 128);
  }
  {  // proc_values: (32768,512) @ (512,128), f32 A converted on load
    int tiles = (32768 / 16) * (128 / 16);
    wmma_gemm_kernel<1, 0, 0><<<(tiles + 7) / 8, 256, 0, stream>>>(
        inputs, Wv_t, (const float*)nullptr, pvw, 32768, 128, 512, 512, 512, 128);
  }

  float* outs_out  = (float*)d_out;
  float* attns_out = outs_out + (size_t)64 * 400 * 160;
  float* stops_out = attns_out + (size_t)64 * 400 * 512;
  decoder_kernel<<<64, 256, 0, stream>>>(
      inputs, mask, pvw, x_all, ak_t, ark_t, attn_b, Wq_t, Wloc_t, v_w, v_b,
      conv_w, proj_t, proj_b, d1k_t, d1rk_t, d1b, d2k_t, d2rk_t, d2b, mel_t,
      mel_b, stop_w, stop_b, outs_out, attns_out, stops_out);
}
